// CausalSelfAttentionALiBi_71116068487606
// MI455X (gfx1250) — matrix operand order repaired and checked by reading, not compiled, not run on hardware
//
#include <hip/hip_runtime.h>
#include <hip/hip_bf16.h>
#include <stdint.h>

typedef __attribute__((ext_vector_type(16))) __bf16 v16bf;
typedef __attribute__((ext_vector_type(8)))  __bf16 v8bf;
typedef __attribute__((ext_vector_type(4)))  __bf16 v4bf;
typedef __attribute__((ext_vector_type(8)))  float  v8f;
typedef __attribute__((ext_vector_type(4)))  float  v4f;

// Problem constants
#define BATCH 4
#define SEQ   2048
#define DMODEL 1024
#define NHEADS 16
#define HDIM  64
#define MROWS (BATCH*SEQ)          // 8192

union Frag { v16bf v; v8bf h[2]; };

__device__ __forceinline__ v8f wmma_bf16(const Frag& a, const Frag& b, v8f c) {
    return __builtin_amdgcn_wmma_f32_16x16x32_bf16(false, a.v, false, b.v,
                                                   (short)0, c, false, false);
}

__device__ __forceinline__ v8f vzero8() {
    v8f z = {0.f,0.f,0.f,0.f,0.f,0.f,0.f,0.f};
    return z;
}

// Async global -> LDS copy, 16B per lane (tracked with ASYNCcnt).
// GV addressing form: vdst = LDS byte offset, vaddr = 64-bit global address.
__device__ __forceinline__ void async_b128(uint32_t lds_off, const __bf16* g) {
    asm volatile("global_load_async_to_lds_b128 %0, %1, off"
                 :: "v"(lds_off), "v"(g) : "memory");
}

// ---------------------------------------------------------------------------
// Cast fp32 -> bf16, 4 elements per thread
// ---------------------------------------------------------------------------
__global__ void cast_f32_to_bf16_x4(const v4f* __restrict__ src,
                                    v4bf* __restrict__ dst, int n4) {
    int i = blockIdx.x * blockDim.x + threadIdx.x;
    int stride = gridDim.x * blockDim.x;
    for (; i < n4; i += stride) {
        v4f x = src[i];
        dst[i] = __builtin_convertvector(x, v4bf);
    }
}

// ---------------------------------------------------------------------------
// QKV projection:  Y = x @ W^T + b  (NT gemm, bf16 WMMA, fp32 accumulate)
// Block: 256 thr (8 waves), tile M=128 x N=64, wave tile 32x32 (2x2 WMMA).
// blockIdx.z selects q / k / v. Writes bf16 (B,H,L,d) q,k and bf16 V^T
// (B,H,d,L); fp32 k,v in natural (B,H,L,d) layout to d_out.
// ---------------------------------------------------------------------------
__global__ __launch_bounds__(256) void qkv_gemm(
    const __bf16* __restrict__ xb,
    const __bf16* __restrict__ wqb, const __bf16* __restrict__ wkb,
    const __bf16* __restrict__ wvb,
    const float* __restrict__ bq, const float* __restrict__ bk,
    const float* __restrict__ bv,
    __bf16* __restrict__ qb, __bf16* __restrict__ kbf, __bf16* __restrict__ vtb,
    float* __restrict__ kout, float* __restrict__ vout)
{
    const int kind = blockIdx.z;
    const __bf16* wb  = (kind == 0) ? wqb : ((kind == 1) ? wkb : wvb);
    const float*  bias = (kind == 0) ? bq  : ((kind == 1) ? bk  : bv);

    const int tid  = threadIdx.x;
    const int wid  = tid >> 5;
    const int lane = tid & 31;
    const int half = lane >> 4;
    const int lr   = lane & 15;

    const int m0 = blockIdx.y * 128 + (wid & 3) * 32;
    const int n0 = blockIdx.x * 64  + (wid >> 2) * 32;

    v8f acc[2][2];
    acc[0][0] = vzero8(); acc[0][1] = vzero8();
    acc[1][0] = vzero8(); acc[1][1] = vzero8();

    const __bf16* aptr[2];
    const __bf16* bptr[2];
#pragma unroll
    for (int am = 0; am < 2; ++am)
        aptr[am] = xb + (size_t)(m0 + am*16 + lr) * DMODEL;
#pragma unroll
    for (int bn = 0; bn < 2; ++bn)
        bptr[bn] = wb + (size_t)(n0 + bn*16 + lr) * DMODEL;

    for (int k = 0; k < DMODEL; k += 32) {
        Frag A[2], Bf[2];
#pragma unroll
        for (int am = 0; am < 2; ++am) {
            const __bf16* p = aptr[am] + k + half*8;
            A[am].h[0] = *(const v8bf*)(p);
            A[am].h[1] = *(const v8bf*)(p + 16);
        }
#pragma unroll
        for (int bn = 0; bn < 2; ++bn) {
            const __bf16* p = bptr[bn] + k + half*8;
            Bf[bn].h[0] = *(const v8bf*)(p);
            Bf[bn].h[1] = *(const v8bf*)(p + 16);
        }
#pragma unroll
        for (int am = 0; am < 2; ++am)
#pragma unroll
            for (int bn = 0; bn < 2; ++bn)
                acc[am][bn] = wmma_bf16(A[am], Bf[bn], acc[am][bn]);
    }

    // Epilogue: bias add + scatter into (B,H,L,d) / (B,H,d,L) layouts
#pragma unroll
    for (int bn = 0; bn < 2; ++bn) {
        const int n  = n0 + bn*16 + lr;
        const float bval = bias[n];
        const int h  = n >> 6;
        const int dd = n & 63;
#pragma unroll
        for (int am = 0; am < 2; ++am) {
#pragma unroll
            for (int r = 0; r < 8; ++r) {
                const int m = m0 + am*16 + r + 8*half;
                const int b = m >> 11;           // / SEQ
                const int l = m & (SEQ - 1);
                const float y = acc[am][bn][r] + bval;
                const size_t idx = ((size_t)(b*NHEADS + h) * SEQ + l) * HDIM + dd;
                if (kind == 0) {
                    qb[idx] = (__bf16)y;
                } else if (kind == 1) {
                    kbf[idx] = (__bf16)y;
                    kout[idx] = y;
                } else {
                    vtb[((size_t)(b*NHEADS + h) * HDIM + dd) * SEQ + l] = (__bf16)y;
                    vout[idx] = y;
                }
            }
        }
    }
}

// ---------------------------------------------------------------------------
// Stage one 64x64 K tile and 64x64 V^T tile into LDS via async copies.
// 128 threads x 4 iterations x 16B = 8KB per tensor.
// ---------------------------------------------------------------------------
__device__ __forceinline__ void stage_tiles(const __bf16* __restrict__ Kb,
                                            const __bf16* __restrict__ Vt,
                                            __bf16 (*Kl)[64], __bf16 (*Vl)[64],
                                            int kt, int tid) {
#pragma unroll
    for (int i = 0; i < 4; ++i) {
        const int c   = tid + i * 128;     // chunk id 0..511
        const int row = c >> 3;            // 64 rows, 8 chunks per row
        const int col = (c & 7) * 8;       // bf16 column
        async_b128((uint32_t)(uintptr_t)&Kl[row][col],
                   Kb + (size_t)(kt * 64 + row) * HDIM + col);
        async_b128((uint32_t)(uintptr_t)&Vl[row][col],
                   Vt + (size_t)row * SEQ + kt * 64 + col);
    }
}

// ---------------------------------------------------------------------------
// FlashAttention (causal + ALiBi). Grid (SEQ/64, B*H), block 128 (4 waves).
// Double-buffered async K/V staging to LDS; each wave owns 16 query rows;
// online softmax; P staged through LDS (C-layout -> A-layout) for P@V.
// ---------------------------------------------------------------------------
__global__ __launch_bounds__(128) void attn_flash(
    const __bf16* __restrict__ qb, const __bf16* __restrict__ kbf,
    const __bf16* __restrict__ vtb, __bf16* __restrict__ ob)
{
    __shared__ __align__(16) __bf16 Kl[2][64][64];    // key tiles (L,d)
    __shared__ __align__(16) __bf16 Vl[2][64][64];    // V^T tiles (d,L)
    __shared__ __align__(16) __bf16 Plds[4][16][64];

    const int tid  = threadIdx.x;
    const int wid  = tid >> 5;
    const int lane = tid & 31;
    const int half = lane >> 4;
    const int lr   = lane & 15;

    const int bh = blockIdx.y;
    const int h  = bh & (NHEADS - 1);
    const int qt = blockIdx.x;
    const int q0 = qt * 64 + wid * 16;

    const float slope = exp2f(-0.5f * (float)(h + 1));
    const float NEGINF = -__builtin_inff();

    const __bf16* Qb = qb  + (size_t)bh * SEQ * HDIM;
    const __bf16* Kb = kbf + (size_t)bh * SEQ * HDIM;
    const __bf16* Vt = vtb + (size_t)bh * HDIM * SEQ;

    // Load Q fragments once (reused for every key tile)
    Frag qf[2];
#pragma unroll
    for (int ks = 0; ks < 2; ++ks) {
        const __bf16* p = Qb + (size_t)(q0 + lr) * HDIM + ks*32 + half*8;
        qf[ks].h[0] = *(const v8bf*)(p);
        qf[ks].h[1] = *(const v8bf*)(p + 16);
    }

    v8f O[4];
#pragma unroll
    for (int dn = 0; dn < 4; ++dn) O[dn] = vzero8();
    float mrow[8], lrow[8];
#pragma unroll
    for (int r = 0; r < 8; ++r) { mrow[r] = NEGINF; lrow[r] = 0.f; }

    // Prime the pipeline with tile 0
    stage_tiles(Kb, Vt, Kl[0], Vl[0], 0, tid);

    for (int kt = 0; kt <= qt; ++kt) {
        const int buf = kt & 1;
        if (kt < qt) {
            // Issue next tile into the other buffer (8 async instrs / wave),
            // then wait until only those 8 remain -> current tile is in LDS.
            stage_tiles(Kb, Vt, Kl[buf ^ 1], Vl[buf ^ 1], kt + 1, tid);
            asm volatile("s_wait_asynccnt 0x8" ::: "memory");
        } else {
            asm volatile("s_wait_asynccnt 0x0" ::: "memory");
        }
        __syncthreads();   // all waves' async portions visible

        // ---- S = Q K^T (16 x 64 per wave), K from LDS ----
        v8f S[4];
#pragma unroll
        for (int an = 0; an < 4; ++an) S[an] = vzero8();
#pragma unroll
        for (int ks = 0; ks < 2; ++ks) {
#pragma unroll
            for (int an = 0; an < 4; ++an) {
                Frag kf;
                const __bf16* p = &Kl[buf][an*16 + lr][ks*32 + half*8];
                kf.h[0] = *(const v8bf*)(p);
                kf.h[1] = *(const v8bf*)(p + 16);
                S[an] = wmma_bf16(qf[ks], kf, S[an]);
            }
        }
        // ---- scale + ALiBi + causal mask + row max ----
        float rmax[8];
#pragma unroll
        for (int r = 0; r < 8; ++r) rmax[r] = NEGINF;
#pragma unroll
        for (int an = 0; an < 4; ++an) {
#pragma unroll
            for (int r = 0; r < 8; ++r) {
                const int q  = q0 + r + 8*half;
                const int kc = kt*64 + an*16 + lr;
                float s = S[an][r] * 0.125f + slope * (float)(kc - q);
                s = (kc <= q) ? s : NEGINF;
                S[an][r] = s;
                rmax[r] = fmaxf(rmax[r], s);
            }
        }
#pragma unroll
        for (int r = 0; r < 8; ++r) {
#pragma unroll
            for (int off = 1; off < 16; off <<= 1)
                rmax[r] = fmaxf(rmax[r], __shfl_xor(rmax[r], off, 32));
        }
        // ---- online softmax rescale ----
        float alpha[8];
#pragma unroll
        for (int r = 0; r < 8; ++r) {
            const float mn = fmaxf(mrow[r], rmax[r]);
            alpha[r] = __expf(mrow[r] - mn);
            mrow[r] = mn;
            lrow[r] *= alpha[r];
        }
#pragma unroll
        for (int dn = 0; dn < 4; ++dn)
#pragma unroll
            for (int r = 0; r < 8; ++r) O[dn][r] *= alpha[r];
        // ---- P = exp(S - m), stage to LDS in row-major [16][64] ----
        float rsum[8];
#pragma unroll
        for (int r = 0; r < 8; ++r) rsum[r] = 0.f;
#pragma unroll
        for (int an = 0; an < 4; ++an) {
#pragma unroll
            for (int r = 0; r < 8; ++r) {
                const float p = __expf(S[an][r] - mrow[r]);
                rsum[r] += p;
                Plds[wid][r + 8*half][an*16 + lr] = (__bf16)p;
            }
        }
#pragma unroll
        for (int r = 0; r < 8; ++r) {
#pragma unroll
            for (int off = 1; off < 16; off <<= 1)
                rsum[r] += __shfl_xor(rsum[r], off, 32);
            lrow[r] += rsum[r];
        }
        // ---- O += P @ V  (A from LDS, B from LDS V^T) ----
#pragma unroll
        for (int ks = 0; ks < 2; ++ks) {
            Frag pf;
            const __bf16* pp = &Plds[wid][lr][ks*32 + half*8];
            pf.h[0] = *(const v8bf*)(pp);
            pf.h[1] = *(const v8bf*)(pp + 16);
#pragma unroll
            for (int dn = 0; dn < 4; ++dn) {
                Frag vf;
                const __bf16* vp = &Vl[buf][dn*16 + lr][ks*32 + half*8];
                vf.h[0] = *(const v8bf*)(vp);
                vf.h[1] = *(const v8bf*)(vp + 16);
                O[dn] = wmma_bf16(pf, vf, O[dn]);
            }
        }
        __syncthreads();   // reads done before this buffer is re-staged
    }

    // ---- finalize: O /= l, write bf16 (B,H,L,d) ----
    float inv[8];
#pragma unroll
    for (int r = 0; r < 8; ++r) inv[r] = 1.0f / lrow[r];
#pragma unroll
    for (int dn = 0; dn < 4; ++dn) {
#pragma unroll
        for (int r = 0; r < 8; ++r) {
            const int q  = q0 + r + 8*half;
            const int dd = dn*16 + lr;
            ob[((size_t)bh * SEQ + q) * HDIM + dd] = (__bf16)(O[dn][r] * inv[r]);
        }
    }
}

// ---------------------------------------------------------------------------
// Output projection: out = o @ Wo^T + bo. A is the (B,H,L,d) attention
// output (head-aware chunk addressing keeps 8-element chunks inside a head).
// ---------------------------------------------------------------------------
__global__ __launch_bounds__(256) void out_gemm(
    const __bf16* __restrict__ ob, const __bf16* __restrict__ wob,
    const float* __restrict__ bo, float* __restrict__ out)
{
    const int tid  = threadIdx.x;
    const int wid  = tid >> 5;
    const int lane = tid & 31;
    const int half = lane >> 4;
    const int lr   = lane & 15;

    const int m0 = blockIdx.y * 128 + (wid & 3) * 32;
    const int n0 = blockIdx.x * 64  + (wid >> 2) * 32;

    v8f acc[2][2];
    acc[0][0] = vzero8(); acc[0][1] = vzero8();
    acc[1][0] = vzero8(); acc[1][1] = vzero8();

    int mb[2], ml[2];
#pragma unroll
    for (int am = 0; am < 2; ++am) {
        const int m = m0 + am*16 + lr;
        mb[am] = m >> 11;
        ml[am] = m & (SEQ - 1);
    }
    const __bf16* bptr[2];
#pragma unroll
    for (int bn = 0; bn < 2; ++bn)
        bptr[bn] = wob + (size_t)(n0 + bn*16 + lr) * DMODEL;

    for (int k = 0; k < DMODEL; k += 32) {
        Frag A[2], Bf[2];
#pragma unroll
        for (int am = 0; am < 2; ++am) {
            const int c0 = k + half*8;
            const int c1 = c0 + 16;
            A[am].h[0] = *(const v8bf*)(ob +
                ((size_t)(mb[am]*NHEADS + (c0 >> 6)) * SEQ + ml[am]) * HDIM + (c0 & 63));
            A[am].h[1] = *(const v8bf*)(ob +
                ((size_t)(mb[am]*NHEADS + (c1 >> 6)) * SEQ + ml[am]) * HDIM + (c1 & 63));
        }
#pragma unroll
        for (int bn = 0; bn < 2; ++bn) {
            const __bf16* p = bptr[bn] + k + half*8;
            Bf[bn].h[0] = *(const v8bf*)(p);
            Bf[bn].h[1] = *(const v8bf*)(p + 16);
        }
#pragma unroll
        for (int am = 0; am < 2; ++am)
#pragma unroll
            for (int bn = 0; bn < 2; ++bn)
                acc[am][bn] = wmma_bf16(A[am], Bf[bn], acc[am][bn]);
    }

#pragma unroll
    for (int bn = 0; bn < 2; ++bn) {
        const int n = n0 + bn*16 + lr;
        const float bval = bo[n];
#pragma unroll
        for (int am = 0; am < 2; ++am) {
#pragma unroll
            for (int r = 0; r < 8; ++r) {
                const int m = m0 + am*16 + r + 8*half;
                out[(size_t)m * DMODEL + n] = acc[am][bn][r] + bval;
            }
        }
    }
}

// ---------------------------------------------------------------------------
// Launch
// ---------------------------------------------------------------------------
extern "C" void kernel_launch(void* const* d_in, const int* in_sizes, int n_in,
                              void* d_out, int out_size, void* d_ws, size_t ws_size,
                              hipStream_t stream) {
    const float* x  = (const float*)d_in[0];
    const float* Wq = (const float*)d_in[1];
    const float* bq = (const float*)d_in[2];
    const float* Wk = (const float*)d_in[3];
    const float* bk = (const float*)d_in[4];
    const float* Wv = (const float*)d_in[5];
    const float* bv = (const float*)d_in[6];
    const float* Wo = (const float*)d_in[7];
    const float* bo = (const float*)d_in[8];
    float* out = (float*)d_out;

    const size_t XB = (size_t)MROWS * DMODEL;      // 8,388,608
    const size_t WN = (size_t)DMODEL * DMODEL;     // 1,048,576

    __bf16* xb  = (__bf16*)d_ws;
    __bf16* wqb = xb  + XB;
    __bf16* wkb = wqb + WN;
    __bf16* wvb = wkb + WN;
    __bf16* wob = wvb + WN;
    __bf16* qb  = wob + WN;
    __bf16* kbf = qb  + XB;
    __bf16* vtb = kbf + XB;
    __bf16* obf = vtb + XB;

    float* kout = out + XB;
    float* vout = out + 2 * XB;

    cast_f32_to_bf16_x4<<<2048, 256, 0, stream>>>((const v4f*)x,  (v4bf*)xb,  (int)(XB/4));
    cast_f32_to_bf16_x4<<<512,  256, 0, stream>>>((const v4f*)Wq, (v4bf*)wqb, (int)(WN/4));
    cast_f32_to_bf16_x4<<<512,  256, 0, stream>>>((const v4f*)Wk, (v4bf*)wkb, (int)(WN/4));
    cast_f32_to_bf16_x4<<<512,  256, 0, stream>>>((const v4f*)Wv, (v4bf*)wvb, (int)(WN/4));
    cast_f32_to_bf16_x4<<<512,  256, 0, stream>>>((const v4f*)Wo, (v4bf*)wob, (int)(WN/4));

    dim3 gQKV(DMODEL/64, MROWS/128, 3);
    qkv_gemm<<<gQKV, 256, 0, stream>>>(xb, wqb, wkb, wvb, bq, bk, bv,
                                       qb, kbf, vtb, kout, vout);

    dim3 gAtt(SEQ/64, BATCH*NHEADS);
    attn_flash<<<gAtt, 128, 0, stream>>>(qb, kbf, vtb, obf);

    dim3 gOut(DMODEL/64, MROWS/128);
    out_gemm<<<gOut, 256, 0, stream>>>(obf, wob, bo, out);
}
